// FaceGeomTransformerLayer_58282706207183
// MI455X (gfx1250) — matrix operand order repaired and checked by reading, not compiled, not run on hardware
//
#include <hip/hip_runtime.h>

// ---------------------------------------------------------------------------
// FaceGeomTransformerLayer for MI455X (gfx1250, wave32, WMMA)
//
// Shapes: BS=4, N=256, DX=256, H=8, DF=32, DFF=1024.
// Roofline: e_add+e_mul = 536 MB streamed once => ~23us floor @ 23.3 TB/s.
// GEMMs total ~1.6 GFLOP => negligible on v_wmma_f32_16x16x32_f16.
// Attention softmax is per-(b,q,h,df) over k => lane-independent online
// softmax. Each lane owns 4 df channels and loads b128 per stream per k-row
// (4x fewer VMEM issues on the hot 536MB stream vs b32); a 64-thread group
// covers one full contiguous 1KB e[b,q,k,:] row.
// ---------------------------------------------------------------------------

constexpr int kBS  = 4;
constexpr int kN   = 256;
constexpr int kDX  = 256;
constexpr int kH   = 8;
constexpr int kDF  = 32;
constexpr int kDFF = 1024;
constexpr int kM   = kBS * kN;   // 1024 rows

typedef __attribute__((ext_vector_type(16))) _Float16 v16h;
typedef __attribute__((ext_vector_type(8)))  _Float16 v8h;
typedef __attribute__((ext_vector_type(4)))  _Float16 v4h;
typedef __attribute__((ext_vector_type(8)))  float    v8f;
typedef __attribute__((ext_vector_type(4)))  float    v4f;

// -------------------------- f32 -> f16 convert ------------------------------
__global__ void cvt_f16_kernel(const float* __restrict__ in,
                               _Float16* __restrict__ out, int n) {
  int i = blockIdx.x * 256 + threadIdx.x;
  if (i < n) out[i] = (_Float16)in[i];
}

// ------------------ f32 -> f16 transpose (Wt[c][k] = W[k][c]) ---------------
__global__ void transpose_cvt_kernel(const float* __restrict__ W,
                                     _Float16* __restrict__ Wt,
                                     int rows, int cols) {
  int c = blockIdx.x * 16 + (threadIdx.x & 15);
  int r = blockIdx.y * 16 + (threadIdx.x >> 4);
  if (c < cols && r < rows)
    Wt[(size_t)c * rows + r] = (_Float16)W[(size_t)r * cols + c];
}

// ----------------------------- WMMA GEMM ------------------------------------
// C[M,Nc] = A[M,Kd](f16,row-major) @ Bt[Nc,Kd](f16,row-major == B^T) + bias
// Optional: per-row mask multiply, ReLU. Writes f32 and/or f16 outputs.
// One wave -> one 16x16 f32 tile; K-loop step 32 via v_wmma_f32_16x16x32_f16.
template <bool RELU>
__global__ void gemm_wmma_kernel(const _Float16* __restrict__ A,
                                 const _Float16* __restrict__ Bt,
                                 const float* __restrict__ bias,
                                 const float* __restrict__ rowmask,
                                 float* __restrict__ Cf,
                                 _Float16* __restrict__ Ch,
                                 int M, int Nc, int Kd) {
  const int lane    = threadIdx.x & 31;
  const int wave    = threadIdx.x >> 5;
  const int colbase = blockIdx.x * 16;
  const int rowbase = (blockIdx.y * 8 + wave) * 16;
  const int l15     = lane & 15;
  const int hi      = lane >> 4;  // 0: lanes 0-15, 1: lanes 16-31

  // A fragment (ISA 16-bit A 16x32 layout): lane<16 holds row l15,
  // K {0..7,16..23}; lane>=16 holds row l15, K {8..15,24..31}.
  const _Float16* ap = A + (size_t)(rowbase + l15) * Kd + hi * 8;
  // B fragment: lane<16 holds column l15, K {0..7,16..23}; lane>=16 K {8..15,24..31}.
  const _Float16* bp = Bt + (size_t)(colbase + l15) * Kd + hi * 8;

  v8f c = {};
  for (int k0 = 0; k0 < Kd; k0 += 32) {
    v8h alo = *(const v8h*)(ap + k0);
    v8h ahi = *(const v8h*)(ap + k0 + 16);
    v8h blo = *(const v8h*)(bp + k0);
    v8h bhi = *(const v8h*)(bp + k0 + 16);
    v16h a = __builtin_shufflevector(alo, ahi, 0, 1, 2, 3, 4, 5, 6, 7,
                                     8, 9, 10, 11, 12, 13, 14, 15);
    v16h b = __builtin_shufflevector(blo, bhi, 0, 1, 2, 3, 4, 5, 6, 7,
                                     8, 9, 10, 11, 12, 13, 14, 15);
    c = __builtin_amdgcn_wmma_f32_16x16x32_f16(
        /*neg_a=*/false, a, /*neg_b=*/false, b,
        /*c_mod=*/(short)0, c, /*reuse_a=*/false, /*reuse_b=*/false);
  }

  const int col  = colbase + l15;
  const float bc = bias ? bias[col] : 0.0f;
#pragma unroll
  for (int j = 0; j < 8; ++j) {
    int row = rowbase + hi * 8 + j;  // C/D layout: VGPR j -> row j (+8 hi half)
    float v = c[j] + bc;
    if (RELU) v = fmaxf(v, 0.0f);
    if (rowmask) v *= rowmask[row];
    if (Cf) Cf[(size_t)row * Nc + col] = v;
    if (Ch) Ch[(size_t)row * Nc + col] = (_Float16)v;
  }
}

// -------------------------- fused attention ---------------------------------
// Block = 4 consecutive (b,q) pairs, 256 threads. 64-thread group per (b,q);
// each lane owns 4 df channels (c0..c0+3) and runs 4 independent online
// softmaxes over k. All hot-stream loads are b128 (16B/lane, 1KB/row/group).
// Output: newXh[b,q,:] = f16( y_add + (y_mul+1) * weighted_V ).
__global__ void attn_kernel(const float* __restrict__ Q,
                            const float* __restrict__ K,
                            const float* __restrict__ V,
                            const float* __restrict__ e_mul,
                            const float* __restrict__ e_add,
                            const float* __restrict__ y_add,
                            const float* __restrict__ y_mul,
                            const float* __restrict__ node_mask,
                            _Float16* __restrict__ newXh) {
  const int grp = threadIdx.x >> 6;            // 0..3: which (b,q) of block
  const int t64 = threadIdx.x & 63;
  const int bq  = blockIdx.x * 4 + grp;        // 0..BS*N-1
  const int b   = bq / kN;
  const int c0  = t64 * 4;                     // channels c0..c0+3

  const float xm_q  = node_mask[bq];
  const float scale = 0.17677669529663687f;    // 1/sqrt(DF)

  v4f qv = *(const v4f*)(Q + (size_t)bq * kDX + c0);
#pragma unroll
  for (int j = 0; j < 4; ++j) qv[j] *= scale;

  const float* em = e_mul + (size_t)bq * kN * kDX + c0;
  const float* ea = e_add + (size_t)bq * kN * kDX + c0;
  const float* Kb = K + (size_t)b * kN * kDX + c0;
  const float* Vb = V + (size_t)b * kN * kDX + c0;
  const float* nm = node_mask + b * kN;

  float m[4], s[4], acc[4];
#pragma unroll
  for (int j = 0; j < 4; ++j) { m[j] = -3.0e38f; s[j] = 0.0f; acc[j] = 0.0f; }

  for (int k = 0; k < kN; ++k) {
    if (k + 8 < kN) {  // keep the two 268MB e-streams deep in flight
      __builtin_prefetch(em + (size_t)(k + 8) * kDX, 0, 1);
      __builtin_prefetch(ea + (size_t)(k + 8) * kDX, 0, 1);
    }
    const float nmk = nm[k];
    const float ee  = xm_q * nmk;
    const v4f em4 = *(const v4f*)(em + (size_t)k * kDX);
    const v4f ea4 = *(const v4f*)(ea + (size_t)k * kDX);
    const v4f kk4 = *(const v4f*)(Kb + (size_t)k * kDX);
    const v4f vv4 = *(const v4f*)(Vb + (size_t)k * kDX);
#pragma unroll
    for (int j = 0; j < 4; ++j) {
      const float e1 = em4[j] * ee;
      const float e2 = ea4[j] * ee;
      float y = qv[j] * kk4[j] * (e1 + 1.0f) + e2;
      y = (nmk > 0.0f) ? y : -3.0e38f;
      const float mn = fmaxf(m[j], y);
      const float so = __expf(m[j] - mn);
      const float p  = __expf(y - mn);
      s[j]   = s[j] * so + p;
      acc[j] = acc[j] * so + p * vv4[j];
      m[j]   = mn;
    }
  }

  const v4f ya = *(const v4f*)(y_add + (size_t)b * kDX + c0);
  const v4f ym = *(const v4f*)(y_mul + (size_t)b * kDX + c0);
  v4h o;
#pragma unroll
  for (int j = 0; j < 4; ++j) {
    const float wv = acc[j] / s[j];
    o[j] = (_Float16)(ya[j] + (ym[j] + 1.0f) * wv);
  }
  *(v4h*)(newXh + (size_t)bq * kDX + c0) = o;  // packed 8B store
}

// ------------------- residual + LayerNorm (block = row) ---------------------
__global__ void ln_kernel(const float* __restrict__ A,
                          const float* __restrict__ B,
                          const float* __restrict__ g,
                          const float* __restrict__ beta,
                          float* __restrict__ outf,
                          _Float16* __restrict__ outh) {
  const int row = blockIdx.x;
  const int t   = threadIdx.x;  // 256 threads == DX
  const float v = A[(size_t)row * kDX + t] + B[(size_t)row * kDX + t];

  float s1 = v, s2 = v * v;
#pragma unroll
  for (int o = 16; o > 0; o >>= 1) {
    s1 += __shfl_down(s1, o);
    s2 += __shfl_down(s2, o);
  }
  __shared__ float r1[8], r2[8];
  __shared__ float mu_s, rstd_s;
  const int w = t >> 5, l = t & 31;
  if (l == 0) { r1[w] = s1; r2[w] = s2; }
  __syncthreads();
  if (t == 0) {
    float S1 = 0.0f, S2 = 0.0f;
    for (int i = 0; i < 8; ++i) { S1 += r1[i]; S2 += r2[i]; }
    const float mu  = S1 / kDX;
    const float var = S2 / kDX - mu * mu;
    mu_s   = mu;
    rstd_s = rsqrtf(var + 1e-5f);
  }
  __syncthreads();
  const float o = (v - mu_s) * rstd_s * g[t] + beta[t];
  outf[(size_t)row * kDX + t] = o;
  if (outh) outh[(size_t)row * kDX + t] = (_Float16)o;
}

// ---------------------------------------------------------------------------
extern "C" void kernel_launch(void* const* d_in, const int* in_sizes, int n_in,
                              void* d_out, int out_size, void* d_ws,
                              size_t ws_size, hipStream_t stream) {
  (void)in_sizes; (void)n_in; (void)out_size; (void)ws_size;

  const float* X     = (const float*)d_in[0];
  const float* e_add = (const float*)d_in[1];
  const float* e_mul = (const float*)d_in[2];
  const float* y_add = (const float*)d_in[3];
  const float* y_mul = (const float*)d_in[4];
  const float* nmask = (const float*)d_in[5];
  const float* Wq = (const float*)d_in[6];  const float* bq = (const float*)d_in[7];
  const float* Wk = (const float*)d_in[8];  const float* bk = (const float*)d_in[9];
  const float* Wv = (const float*)d_in[10]; const float* bv = (const float*)d_in[11];
  const float* Wo = (const float*)d_in[12]; const float* bo = (const float*)d_in[13];
  const float* W1 = (const float*)d_in[14]; const float* b1 = (const float*)d_in[15];
  const float* W2 = (const float*)d_in[16]; const float* b2 = (const float*)d_in[17];
  const float* g1 = (const float*)d_in[18]; const float* be1 = (const float*)d_in[19];
  const float* g2 = (const float*)d_in[20]; const float* be2 = (const float*)d_in[21];
  float* X2 = (float*)d_out;

  // workspace carve (256B aligned)
  char* p = (char*)d_ws;
  auto alloc = [&](size_t bytes) -> void* {
    void* r = (void*)p;
    p += (bytes + 255) & ~(size_t)255;
    return r;
  };
  _Float16* Xh    = (_Float16*)alloc((size_t)kM * kDX * 2);
  _Float16* Wqt   = (_Float16*)alloc((size_t)kDX * kDX * 2);
  _Float16* Wkt   = (_Float16*)alloc((size_t)kDX * kDX * 2);
  _Float16* Wvt   = (_Float16*)alloc((size_t)kDX * kDX * 2);
  _Float16* Wot   = (_Float16*)alloc((size_t)kDX * kDX * 2);
  _Float16* W1t   = (_Float16*)alloc((size_t)kDFF * kDX * 2);
  _Float16* W2t   = (_Float16*)alloc((size_t)kDX * kDFF * 2);
  float*    Qf    = (float*)alloc((size_t)kM * kDX * 4);
  float*    Kf    = (float*)alloc((size_t)kM * kDX * 4);
  float*    Vf    = (float*)alloc((size_t)kM * kDX * 4);
  _Float16* newXh = (_Float16*)alloc((size_t)kM * kDX * 2);
  float*    proj  = (float*)alloc((size_t)kM * kDX * 4);
  float*    X1f   = (float*)alloc((size_t)kM * kDX * 4);
  _Float16* X1h   = (_Float16*)alloc((size_t)kM * kDX * 2);
  _Float16* Hh    = (_Float16*)alloc((size_t)kM * kDFF * 2);
  float*    ffo   = (float*)alloc((size_t)kM * kDX * 4);

  const dim3 b256(256);

  // stage f16 operands
  cvt_f16_kernel<<<(kM * kDX) / 256, b256, 0, stream>>>(X, Xh, kM * kDX);
  transpose_cvt_kernel<<<dim3(kDX / 16, kDX / 16), b256, 0, stream>>>(Wq, Wqt, kDX, kDX);
  transpose_cvt_kernel<<<dim3(kDX / 16, kDX / 16), b256, 0, stream>>>(Wk, Wkt, kDX, kDX);
  transpose_cvt_kernel<<<dim3(kDX / 16, kDX / 16), b256, 0, stream>>>(Wv, Wvt, kDX, kDX);
  transpose_cvt_kernel<<<dim3(kDX / 16, kDX / 16), b256, 0, stream>>>(Wo, Wot, kDX, kDX);
  transpose_cvt_kernel<<<dim3(kDFF / 16, kDX / 16), b256, 0, stream>>>(W1, W1t, kDX, kDFF);
  transpose_cvt_kernel<<<dim3(kDX / 16, kDFF / 16), b256, 0, stream>>>(W2, W2t, kDFF, kDX);

  // QKV projections (bias + x_mask fused)
  const dim3 gdx(kDX / 16, kM / 128);
  gemm_wmma_kernel<false><<<gdx, b256, 0, stream>>>(Xh, Wqt, bq, nmask, Qf, nullptr, kM, kDX, kDX);
  gemm_wmma_kernel<false><<<gdx, b256, 0, stream>>>(Xh, Wkt, bk, nmask, Kf, nullptr, kM, kDX, kDX);
  gemm_wmma_kernel<false><<<gdx, b256, 0, stream>>>(Xh, Wvt, bv, nmask, Vf, nullptr, kM, kDX, kDX);

  // fused per-feature attention + y_x modulation (streams 536MB of e)
  attn_kernel<<<kM / 4, b256, 0, stream>>>(Qf, Kf, Vf, e_mul, e_add, y_add, y_mul, nmask, newXh);

  // output projection (masked), residual + LN1
  gemm_wmma_kernel<false><<<gdx, b256, 0, stream>>>(newXh, Wot, bo, nmask, proj, nullptr, kM, kDX, kDX);
  ln_kernel<<<kM, b256, 0, stream>>>(X, proj, g1, be1, X1f, X1h);

  // FFN: relu(X1 @ W1 + b1) @ W2 + b2, residual + LN2
  const dim3 gff(kDFF / 16, kM / 128);
  gemm_wmma_kernel<true><<<gff, b256, 0, stream>>>(X1h, W1t, b1, nullptr, nullptr, Hh, kM, kDFF, kDX);
  gemm_wmma_kernel<false><<<gdx, b256, 0, stream>>>(Hh, W2t, b2, nullptr, ffo, nullptr, kM, kDX, kDFF);
  ln_kernel<<<kM, b256, 0, stream>>>(X1f, ffo, g2, be2, X2, nullptr);
}
